// IterBlock_10668698763289
// MI455X (gfx1250) — matrix operand order repaired and checked by reading, not compiled, not run on hardware
//
#include <hip/hip_runtime.h>
#include <math.h>

// ---------------------------------------------------------------------------
// MI455X (gfx1250) IterBlock implementation.
// All contractions go through v_wmma_f32_16x16x32_f16 (f16 in, f32 accumulate).
// Residual streams (msa, pair) kept f32; GEMM operands staged as f16.
// ---------------------------------------------------------------------------

typedef __attribute__((ext_vector_type(16))) _Float16 v16h;
typedef __attribute__((ext_vector_type(8)))  _Float16 v8h;
typedef __attribute__((ext_vector_type(8)))  float    v8f;

#define TM 128
#define TN 64
#define TK 32
#define LDS_PAD 8   // row stride 40 halves = 80 B: 16B-aligned, conflict-free

static __device__ __forceinline__ v16h cat8(v8h a, v8h b) {
  return __builtin_shufflevector(a, b, 0, 1, 2, 3, 4, 5, 6, 7,
                                 8, 9, 10, 11, 12, 13, 14, 15);
}

// ---------------------------------------------------------------------------
// Batched, strided WMMA GEMM:  C[M,N] = act(alpha * A[M,K] @ Bt[N,K]^T + bias)
// A, Bt are f16 (Bt is B transposed so both stream contiguously along K).
// C (f32, optional) and C16 (f16 mirror, optional) share ldc. 8 waves/block;
// each wave owns a 32x32 tile = 2x2 v_wmma_f32_16x16x32_f16 accumulators.
// Fragments are built with 128-bit LDS vector loads that match the CDNA5
// VGPR layouts exactly (CDNA5 ISA §7.12.2) — lane-half h holds K = 8h..8h+7 and 16+8h..16+8h+7,
// so no per-element shuffling (and no scratch) is needed.
// Requires K % 32 == 0 and 16-byte aligned A/Bt rows (guaranteed by caller).
// ---------------------------------------------------------------------------
__global__ __launch_bounds__(256)
void wmma_gemm_kernel(const _Float16* __restrict__ A, const _Float16* __restrict__ Bt,
                      const float* __restrict__ bias,
                      float* __restrict__ C, _Float16* __restrict__ C16,
                      int M, int N, int K, int lda, int ldb, int ldc,
                      long sA, long sB, long sC,
                      float alpha, int act, int accum)
{
  __shared__ _Float16 As[TM][TK + LDS_PAD];
  __shared__ _Float16 Bs[TN][TK + LDS_PAD];
  const int batch = blockIdx.z;
  A  += (long)batch * sA;
  Bt += (long)batch * sB;
  const long cb = (long)batch * sC;
  const int m0 = blockIdx.y * TM;
  const int n0 = blockIdx.x * TN;
  const int tid  = threadIdx.x;
  const int wave = tid >> 5, lane = tid & 31;
  const int wm = (wave & 3) * 32, wn = (wave >> 2) * 32;
  const int lh = lane >> 4, lr = lane & 15;

  v8f acc[2][2] = {};

  for (int k0 = 0; k0 < K; k0 += TK) {
    // Stage A tile (128x32) and Bt tile (64x32) with 128-bit transfers.
    // TM*TK/8 = 512 vectors -> 2 per thread; TN*TK/8 = 256 -> 1 per thread.
    for (int i = tid; i < TM * (TK / 8); i += 256) {
      int r = i >> 2, c = (i & 3) << 3;
      int gm = m0 + r, gk = k0 + c;
      v8h v = {};
      if (gm < M && gk < K) v = *(const v8h*)(A + (long)gm * lda + gk);
      *(v8h*)&As[r][c] = v;
    }
    for (int i = tid; i < TN * (TK / 8); i += 256) {
      int r = i >> 2, c = (i & 3) << 3;
      int gn = n0 + r, gk = k0 + c;
      v8h v = {};
      if (gn < N && gk < K) v = *(const v8h*)(Bt + (long)gn * ldb + gk);
      *(v8h*)&Bs[r][c] = v;
    }
    // Prefetch next K tile into cache (global_prefetch_b8 on gfx1250).
    if (k0 + TK < K) {
      int gm = m0 + (tid >> 1);
      if (gm < M) __builtin_prefetch(A + (long)gm * lda + k0 + TK + (tid & 1) * 16, 0, 1);
      int gn = n0 + (tid & 63);
      if (gn < N) __builtin_prefetch(Bt + (long)gn * ldb + k0 + TK, 0, 1);
    }
    __syncthreads();

    // A fragment: lane-half lh holds K = 8*lh..8*lh+7 (regs 0-3) and
    // K = 16+8*lh.. (regs 4-7): two contiguous b128 LDS loads.
    v16h af[2], bf[2];
#pragma unroll
    for (int mt = 0; mt < 2; ++mt) {
      const _Float16* ap = &As[wm + mt * 16 + lr][8 * lh];
      af[mt] = cat8(*(const v8h*)ap, *(const v8h*)(ap + 16));
    }
    // B fragment: same K order as A (8*lh.. and 16+8*lh..).
#pragma unroll
    for (int nt = 0; nt < 2; ++nt) {
      const _Float16* bp = &Bs[wn + nt * 16 + lr][8 * lh];
      bf[nt] = cat8(*(const v8h*)bp, *(const v8h*)(bp + 16));
    }
#pragma unroll
    for (int mt = 0; mt < 2; ++mt)
#pragma unroll
      for (int nt = 0; nt < 2; ++nt)
        acc[mt][nt] = __builtin_amdgcn_wmma_f32_16x16x32_f16(
            false, af[mt], false, bf[nt], (short)0, acc[mt][nt], false, false);
    __syncthreads();
  }

  // Epilogue per f32 C/D layout: VGPR r -> M = r (+8 for lanes 16..31), N = lane&15.
  for (int mt = 0; mt < 2; ++mt)
    for (int nt = 0; nt < 2; ++nt)
#pragma unroll
      for (int r = 0; r < 8; ++r) {
        int row = m0 + wm + mt * 16 + r + lh * 8;
        int col = n0 + wn + nt * 16 + lr;
        if (row < M && col < N) {
          float v = acc[mt][nt][r] * alpha;
          if (bias) v += bias[col];
          long idx = cb + (long)row * ldc + col;
          if (accum && C) v += C[idx];
          if (act == 1)      v = v > 0.f ? v : 0.f;
          else if (act == 2) v = 1.f / (1.f + __expf(-v));
          if (C)   C[idx]   = v;
          if (C16) C16[idx] = (_Float16)v;
        }
      }
}

// ---------------------------------------------------------------------------
// LayerNorm over last dim, one wave32 per row. Optional f32 / f16 outputs.
// ---------------------------------------------------------------------------
__global__ __launch_bounds__(256)
void layernorm_kernel(const float* __restrict__ x, const float* __restrict__ g,
                      const float* __restrict__ b, float* __restrict__ y32,
                      _Float16* __restrict__ y16, long rows, int dim)
{
  long row = (long)blockIdx.x * 8 + (threadIdx.x >> 5);
  if (row >= rows) return;
  int lane = threadIdx.x & 31;
  const float* xr = x + row * dim;
  float s = 0.f, s2 = 0.f;
  for (int d = lane; d < dim; d += 32) { float v = xr[d]; s += v; s2 += v * v; }
  for (int m = 16; m; m >>= 1) { s += __shfl_xor(s, m); s2 += __shfl_xor(s2, m); }
  float mean = s / dim;
  float inv  = rsqrtf(s2 / dim - mean * mean + 1e-5f);
  for (int d = lane; d < dim; d += 32) {
    float v = (xr[d] - mean) * inv * g[d] + b[d];
    if (y32) y32[row * dim + d] = v;
    if (y16) y16[row * dim + d] = (_Float16)v;
  }
}

// ---------------------------------------------------------------------------
// Elementwise: 0:y=a  1:y=a+b  2:y=a*b  3:y=sigmoid(a)*b  4:y+=a*b
// ---------------------------------------------------------------------------
__global__ __launch_bounds__(256)
void ew_kernel(const float* __restrict__ a, const float* __restrict__ b,
               float* __restrict__ y32, _Float16* __restrict__ y16, long n, int mode)
{
  long i = (long)blockIdx.x * 256 + threadIdx.x;
  if (i >= n) return;
  float va = a ? a[i] : 0.f;
  float vb = b ? b[i] : 0.f;
  float v;
  switch (mode) {
    case 1:  v = va + vb; break;
    case 2:  v = va * vb; break;
    case 3:  v = vb / (1.f + __expf(-va)); break;
    case 4:  v = y32[i] + va * vb; break;
    default: v = va; break;
  }
  if (y32) y32[i] = v;
  if (y16) y16[i] = (_Float16)v;
}

// ---------------------------------------------------------------------------
// Generic strided 4D permutes (head packing / transposes).
// ---------------------------------------------------------------------------
__global__ __launch_bounds__(256)
void permute4_f16_kernel(const _Float16* __restrict__ x, _Float16* __restrict__ y,
                         int d1, int d2, int d3,
                         long s0, long s1, long s2, long s3,
                         long t0, long t1, long t2, long t3, long n)
{
  long i = (long)blockIdx.x * 256 + threadIdx.x;
  if (i >= n) return;
  int i3 = (int)(i % d3); long r = i / d3;
  int i2 = (int)(r % d2); r /= d2;
  int i1 = (int)(r % d1); long i0 = r / d1;
  y[i0 * t0 + i1 * t1 + i2 * t2 + i3 * t3] = x[i0 * s0 + i1 * s1 + i2 * s2 + i3 * s3];
}

__global__ __launch_bounds__(256)
void permute4_f32f16_kernel(const float* __restrict__ x, _Float16* __restrict__ y,
                            int d1, int d2, int d3,
                            long s0, long s1, long s2, long s3,
                            long t0, long t1, long t2, long t3, long n)
{
  long i = (long)blockIdx.x * 256 + threadIdx.x;
  if (i >= n) return;
  int i3 = (int)(i % d3); long r = i / d3;
  int i2 = (int)(r % d2); r /= d2;
  int i1 = (int)(r % d1); long i0 = r / d1;
  y[i0 * t0 + i1 * t1 + i2 * t2 + i3 * t3] =
      (_Float16)x[i0 * s0 + i1 * s1 + i2 * s2 + i3 * s3];
}

__global__ __launch_bounds__(256)
void permute4_f32_kernel(const float* __restrict__ x, float* __restrict__ y,
                         int d1, int d2, int d3,
                         long s0, long s1, long s2, long s3,
                         long t0, long t1, long t2, long t3, long n, int accum)
{
  long i = (long)blockIdx.x * 256 + threadIdx.x;
  if (i >= n) return;
  int i3 = (int)(i % d3); long r = i / d3;
  int i2 = (int)(r % d2); r /= d2;
  int i1 = (int)(r % d1); long i0 = r / d1;
  long dst = i0 * t0 + i1 * t1 + i2 * t2 + i3 * t3;
  float v = x[i0 * s0 + i1 * s1 + i2 * s2 + i3 * s3];
  y[dst] = accum ? (y[dst] + v) : v;
}

// ---------------------------------------------------------------------------
// Row softmax (wave per row). Optional f16 mirror.
// ---------------------------------------------------------------------------
__global__ __launch_bounds__(256)
void softmax_kernel(float* __restrict__ x, _Float16* __restrict__ y16, long rows, int cols)
{
  long row = (long)blockIdx.x * 8 + (threadIdx.x >> 5);
  if (row >= rows) return;
  int lane = threadIdx.x & 31;
  float* xr = x + row * cols;
  float mx = -3.0e38f;
  for (int c = lane; c < cols; c += 32) mx = fmaxf(mx, xr[c]);
  for (int m = 16; m; m >>= 1) mx = fmaxf(mx, __shfl_xor(mx, m));
  float s = 0.f;
  for (int c = lane; c < cols; c += 32) { float e = __expf(xr[c] - mx); xr[c] = e; s += e; }
  for (int m = 16; m; m >>= 1) s += __shfl_xor(s, m);
  float inv = 1.f / s;
  for (int c = lane; c < cols; c += 32) {
    float v = xr[c] * inv;
    xr[c] = v;
    if (y16) y16[row * cols + c] = (_Float16)v;
  }
}

// logits[h, a, j] + bias[(a,j) or (j,a), h] -> softmax over j, f16 out [h,a,j]
__global__ __launch_bounds__(256)
void softmax_bias_kernel(float* __restrict__ lg, const float* __restrict__ bias,
                         _Float16* __restrict__ y16, int H, int L, int tb)
{
  long row = (long)blockIdx.x * 8 + (threadIdx.x >> 5);
  if (row >= (long)H * L) return;
  int h = (int)(row / L), a = (int)(row % L);
  int lane = threadIdx.x & 31;
  float* xr = lg + row * L;
  float mx = -3.0e38f;
  for (int j = lane; j < L; j += 32) {
    long bo = tb ? ((long)j * L + a) : ((long)a * L + j);
    float v = xr[j] + bias[bo * H + h];
    xr[j] = v;
    mx = fmaxf(mx, v);
  }
  for (int m = 16; m; m >>= 1) mx = fmaxf(mx, __shfl_xor(mx, m));
  float s = 0.f;
  for (int j = lane; j < L; j += 32) { float e = __expf(xr[j] - mx); xr[j] = e; s += e; }
  for (int m = 16; m; m >>= 1) s += __shfl_xor(s, m);
  float inv = 1.f / s;
  for (int j = lane; j < L; j += 32) y16[row * L + j] = (_Float16)(xr[j] * inv);
}

// ---------------------------------------------------------------------------
// Tied sequence-weight logits: sl[(l,h), s] = q0[l,h,:] . k0[(s,l),h,:]
// ---------------------------------------------------------------------------
__global__ __launch_bounds__(256)
void seqweight_kernel(const float* __restrict__ q0, const float* __restrict__ k0,
                      float* __restrict__ sl, int S, int L, int H, int Dh)
{
  long i = (long)blockIdx.x * 256 + threadIdx.x;
  if (i >= (long)L * H * S) return;
  int s = (int)(i % S); long r = i / S;
  int h = (int)(r % H); int l = (int)(r / H);
  const float* qp = q0 + (long)l * H * Dh + h * Dh;
  const float* kp = k0 + ((long)s * L + l) * (long)H * Dh + h * Dh;
  float acc = 0.f;
  for (int d = 0; d < Dh; ++d) acc += qp[d] * kp[d];
  sl[((long)l * H + h) * S + s] = acc;
}

// q16[s,l,h,d] = q32[s,l,h,d] * sw[(l,h), s]
__global__ __launch_bounds__(256)
void mul_sw_kernel(const float* __restrict__ q32, const float* __restrict__ sw,
                   _Float16* __restrict__ q16, int S, int L, int H, int Dh, long n)
{
  long i = (long)blockIdx.x * 256 + threadIdx.x;
  if (i >= n) return;
  long r = i / Dh;
  int h = (int)(r % H); r /= H;
  int l = (int)(r % L); int s = (int)(r / L);
  q16[i] = (_Float16)(q32[i] * sw[((long)l * H + h) * S + s]);
}

// y[(l,m), i*Dhs+j] = a[l,i] * b[m,j]
__global__ __launch_bounds__(256)
void outer16_kernel(const float* __restrict__ a, const float* __restrict__ b,
                    _Float16* __restrict__ y, int L, int Dhs, long n)
{
  long i = (long)blockIdx.x * 256 + threadIdx.x;
  if (i >= n) return;
  int j = (int)(i % Dhs); long r = i / Dhs;
  int ii = (int)(r % Dhs); r /= Dhs;
  int m = (int)(r % L); int l = (int)(r / L);
  y[i] = (_Float16)(a[(long)l * Dhs + ii] * b[(long)m * Dhs + j]);
}

// ---------------------------------------------------------------------------
// Host orchestration
// ---------------------------------------------------------------------------
namespace {
constexpr int SQ  = 128;   // N sequences
constexpr int LQ  = 256;   // L residues
constexpr int DM  = 256;   // d_msa
constexpr int DP  = 128;   // d_pair
constexpr int DS  = 32;    // d_state
constexpr int DR  = 64;    // d_rbf
constexpr int HM  = 8, HP = 4, DHD = 32, DHS = 16;
constexpr long TOKM = (long)SQ * LQ;   // 32768 msa tokens
constexpr long TOKP = (long)LQ * LQ;   // 65536 pair tokens

// flattened setup_inputs() order
enum {
  IN_MSA = 0, IN_PAIR, IN_RBF, IN_STATE,
  M2M_NP_G = 4, M2M_NP_B, M2M_ERBF_W, M2M_ERBF_B, M2M_NS_G, M2M_NS_B, M2M_PS_W, M2M_PS_B,
  ROW_NM_G = 12, ROW_NM_B, ROW_Q, ROW_K, ROW_V, ROW_GW, ROW_GB, ROW_OW, ROW_OB,
  ROW_NP_G = 21, ROW_NP_B, ROW_B, ROW_SWQ, ROW_SWQB, ROW_SWK, ROW_SWKB,
  COL_NM_G = 28, COL_NM_B, COL_Q, COL_K, COL_V, COL_GW, COL_GB, COL_OW, COL_OB,
  MFF_NG = 37, MFF_NB, MFF_W1, MFF_B1, MFF_W2, MFF_B2,
  M2P_NG = 43, M2P_NB, M2P_PLW, M2P_PLB, M2P_PRW, M2P_PRB, M2P_POW, M2P_POB,
  P2P_NSG = 51, P2P_NSB, P2P_PLW, P2P_PLB, P2P_PRW, P2P_PRB, P2P_TGW, P2P_TGB,
  P2P_ERBFW, P2P_ERBFB,
  TO_NG = 61,  // tri_out: n_g,n_b,lp_w,lp_b,rp_w,rp_b,lg_w,lg_b,rg_w,rg_b,g_w,g_b,no_g,no_b,op_w,op_b
  TI_NG = 77,
  PR_NM_G = 93,   // pair row attn: nm_g,nm_b,q,k,v,g_w,g_b,o_w,o_b,np_g,np_b,b
  PC_NM_G = 105,
  PFF_NG = 117, PFF_NB, PFF_W1, PFF_B1, PFF_W2, PFF_B2
};
} // namespace

extern "C" void kernel_launch(void* const* d_in, const int* in_sizes, int n_in,
                              void* d_out, int out_size, void* d_ws, size_t ws_size,
                              hipStream_t stream)
{
  (void)in_sizes; (void)n_in; (void)out_size; (void)ws_size;
  const float DHI = 0.17677669529663689f;  // DH^-0.5

  // ----- workspace arena -----
  char* wsb = (char*)d_ws;
  size_t off = 0;
  auto alloc = [&](size_t bytes) -> void* {
    off = (off + 255) & ~(size_t)255;
    void* p = wsb + off; off += bytes; return p;
  };
  float*     MSA  = (float*)alloc(TOKM * DM * 4);
  float*     PAIR = (float*)alloc(TOKP * DP * 4);
  float*     F0   = (float*)alloc(TOKP * DP * 4);
  float*     F1   = (float*)alloc(TOKP * DP * 4);
  float*     F2   = (float*)alloc(TOKP * DP * 4);
  float*     F3   = (float*)alloc(TOKP * DP * 4);
  _Float16*  H0   = (_Float16*)alloc(TOKM * 1024 * 2);
  _Float16*  H1   = (_Float16*)alloc(TOKM * DM * 2);
  _Float16*  H2   = (_Float16*)alloc(TOKM * DM * 2);
  _Float16*  H3   = (_Float16*)alloc(TOKM * DM * 2);
  _Float16*  H4   = (_Float16*)alloc(TOKM * DM * 2);
  float*     G    = (float*)alloc((size_t)4 * 1024 * 1024 * 4);
  _Float16*  SML  = (_Float16*)alloc(1 << 20);
  float* Gbias = G;            // 65536*8
  float* Gsl   = G + 524288;   // 2048*128
  float* Gq0   = G + 786432;   // 256*256
  float* Gst   = G + 851968;   // 256*256
  float* Gm    = G + 917504;   // misc (~3.2M floats)

  // ----- helpers -----
  auto BP = [&](int i) { return (const float*)d_in[i]; };
  auto GEMM = [&](const _Float16* A, const _Float16* Bt, const float* bias,
                  float* C, _Float16* C16, int M, int N, int K,
                  int lda, int ldb, int ldc, long sA, long sB, long sC, int batch,
                  float alpha, int act, int accum) {
    wmma_gemm_kernel<<<dim3((N + TN - 1) / TN, (M + TM - 1) / TM, batch),
                       dim3(256), 0, stream>>>(
        A, Bt, bias, C, C16, M, N, K, lda, ldb, ldc, sA, sB, sC, alpha, act, accum);
  };
  auto LN = [&](const float* x, int gi, int bi, float* y32, _Float16* y16,
                long rows, int dim) {
    layernorm_kernel<<<dim3((unsigned)((rows + 7) / 8)), dim3(256), 0, stream>>>(
        x, BP(gi), BP(bi), y32, y16, rows, dim);
  };
  auto EW = [&](const float* a, const float* b, float* y32, _Float16* y16,
                long n, int mode) {
    ew_kernel<<<dim3((unsigned)((n + 255) / 256)), dim3(256), 0, stream>>>(a, b, y32, y16, n, mode);
  };
  auto P16 = [&](const _Float16* x, _Float16* y, int d0, int d1, int d2, int d3,
                 long s0, long s1, long s2, long s3,
                 long t0, long t1, long t2, long t3) {
    long n = (long)d0 * d1 * d2 * d3;
    permute4_f16_kernel<<<dim3((unsigned)((n + 255) / 256)), dim3(256), 0, stream>>>(
        x, y, d1, d2, d3, s0, s1, s2, s3, t0, t1, t2, t3, n);
  };
  auto P3216 = [&](const float* x, _Float16* y, int d0, int d1, int d2, int d3,
                   long s0, long s1, long s2, long s3,
                   long t0, long t1, long t2, long t3) {
    long n = (long)d0 * d1 * d2 * d3;
    permute4_f32f16_kernel<<<dim3((unsigned)((n + 255) / 256)), dim3(256), 0, stream>>>(
        x, y, d1, d2, d3, s0, s1, s2, s3, t0, t1, t2, t3, n);
  };
  auto P32 = [&](const float* x, float* y, int d0, int d1, int d2, int d3,
                 long s0, long s1, long s2, long s3,
                 long t0, long t1, long t2, long t3, int accum) {
    long n = (long)d0 * d1 * d2 * d3;
    permute4_f32_kernel<<<dim3((unsigned)((n + 255) / 256)), dim3(256), 0, stream>>>(
        x, y, d1, d2, d3, s0, s1, s2, s3, t0, t1, t2, t3, n, accum);
  };
  auto SM = [&](float* x, _Float16* y16, long rows, int cols) {
    softmax_kernel<<<dim3((unsigned)((rows + 7) / 8)), dim3(256), 0, stream>>>(x, y16, rows, cols);
  };
  // weights: f32 [din,dout] -> f16 [dout,din] so GEMM streams K contiguously
  auto PACKW = [&](int wi, int din, int dout) -> _Float16* {
    _Float16* wt = (_Float16*)alloc((size_t)din * dout * 2);
    P3216(BP(wi), wt, dout, din, 1, 1, 1, dout, 0, 0, din, 1, 0, 0);
    return wt;
  };

  // ----- pack all GEMM weights to f16 (recomputed every call; deterministic) -----
  _Float16 *WrowQ = PACKW(ROW_Q, DM, DM), *WrowK = PACKW(ROW_K, DM, DM),
           *WrowV = PACKW(ROW_V, DM, DM), *WrowG = PACKW(ROW_GW, DM, DM),
           *WrowO = PACKW(ROW_OW, DM, DM),
           *WswQ  = PACKW(ROW_SWQ, DM, DM), *WswK = PACKW(ROW_SWK, DM, DM),
           *WrowB = PACKW(ROW_B, DP, HM);
  _Float16 *Werbf = PACKW(M2M_ERBF_W, DR, DP), *Wps = PACKW(M2M_PS_W, DS, DM);
  _Float16 *WcolQ = PACKW(COL_Q, DM, DM), *WcolK = PACKW(COL_K, DM, DM),
           *WcolV = PACKW(COL_V, DM, DM), *WcolG = PACKW(COL_GW, DM, DM),
           *WcolO = PACKW(COL_OW, DM, DM);
  _Float16 *Wff1 = PACKW(MFF_W1, DM, 4 * DM), *Wff2 = PACKW(MFF_W2, 4 * DM, DM);
  _Float16 *WplM = PACKW(M2P_PLW, DM, DHD), *WprM = PACKW(M2P_PRW, DM, DHD),
           *WpoM = PACKW(M2P_POW, DHD * DHD, DP);
  _Float16 *Wp2L = PACKW(P2P_PLW, DS, DHS), *Wp2R = PACKW(P2P_PRW, DS, DHS),
           *Wtg  = PACKW(P2P_TGW, DHS * DHS, DP), *Werbf2 = PACKW(P2P_ERBFW, DR, DP);
  _Float16 *WtoLP = PACKW(TO_NG + 2, DP, DHD), *WtoRP = PACKW(TO_NG + 4, DP, DHD),
           *WtoLG = PACKW(TO_NG + 6, DP, DHD), *WtoRG = PACKW(TO_NG + 8, DP, DHD),
           *WtoG  = PACKW(TO_NG + 10, DP, DP), *WtoOP = PACKW(TO_NG + 14, DHD, DP);
  _Float16 *WtiLP = PACKW(TI_NG + 2, DP, DHD), *WtiRP = PACKW(TI_NG + 4, DP, DHD),
           *WtiLG = PACKW(TI_NG + 6, DP, DHD), *WtiRG = PACKW(TI_NG + 8, DP, DHD),
           *WtiG  = PACKW(TI_NG + 10, DP, DP), *WtiOP = PACKW(TI_NG + 14, DHD, DP);
  _Float16 *WprQ2 = PACKW(PR_NM_G + 2, DP, DP), *WprK2 = PACKW(PR_NM_G + 3, DP, DP),
           *WprV2 = PACKW(PR_NM_G + 4, DP, DP), *WprG2 = PACKW(PR_NM_G + 5, DP, DP),
           *WprO2 = PACKW(PR_NM_G + 7, DP, DP), *WprB2 = PACKW(PR_NM_G + 11, DP, HP);
  _Float16 *WpcQ2 = PACKW(PC_NM_G + 2, DP, DP), *WpcK2 = PACKW(PC_NM_G + 3, DP, DP),
           *WpcV2 = PACKW(PC_NM_G + 4, DP, DP), *WpcG2 = PACKW(PC_NM_G + 5, DP, DP),
           *WpcO2 = PACKW(PC_NM_G + 7, DP, DP), *WpcB2 = PACKW(PC_NM_G + 11, DP, HP);
  _Float16 *Wpf1 = PACKW(PFF_W1, DP, 2 * DP), *Wpf2 = PACKW(PFF_W2, 2 * DP, DP);

  // ================= Phase 1: msa2msa =================
  hipMemcpyAsync(MSA, d_in[IN_MSA], TOKM * DM * sizeof(float), hipMemcpyDeviceToDevice, stream);
  hipMemcpyAsync(PAIR, d_in[IN_PAIR], TOKP * DP * sizeof(float), hipMemcpyDeviceToDevice, stream);

  // pr = LN(pair) + rbf @ erbf_w + erbf_b
  LN((const float*)d_in[IN_PAIR], M2M_NP_G, M2M_NP_B, F0, nullptr, TOKP, DP);
  EW((const float*)d_in[IN_RBF], nullptr, nullptr, H4, TOKP * DR, 0);  // rbf -> f16
  GEMM(H4, Werbf, BP(M2M_ERBF_B), F0, nullptr, (int)TOKP, DP, DR, DR, DR, DP,
       0, 0, 0, 1, 1.f, 0, 1);
  // pair bias for row attention: LN(pr) @ b  -> Gbias [(q,k), H]
  LN(F0, ROW_NP_G, ROW_NP_B, nullptr, H2, TOKP, DP);
  GEMM(H2, WrowB, nullptr, Gbias, nullptr, (int)TOKP, HM, DP, DP, DP, HM,
       0, 0, 0, 1, 1.f, 0, 0);
  // state injection into msa[:,0]
  LN((const float*)d_in[IN_STATE], M2M_NS_G, M2M_NS_B, nullptr, SML, LQ, DS);
  GEMM(SML, Wps, BP(M2M_PS_B), Gst, nullptr, LQ, DM, DS, DS, DS, DM, 0, 0, 0, 1, 1.f, 0, 0);
  EW(MSA, Gst, MSA, nullptr, (long)LQ * DM, 1);

  // ---- tied row attention ----
  LN(MSA, ROW_NM_G, ROW_NM_B, nullptr, H1, TOKM, DM);
  GEMM(H1, WswQ, BP(ROW_SWQB), Gq0, nullptr, LQ, DM, DM, DM, DM, DM, 0, 0, 0, 1, DHI, 0, 0);
  GEMM(H1, WswK, BP(ROW_SWKB), F2, nullptr, (int)TOKM, DM, DM, DM, DM, DM, 0, 0, 0, 1, 1.f, 0, 0);
  seqweight_kernel<<<dim3((unsigned)(((long)LQ * HM * SQ + 255) / 256)), dim3(256), 0, stream>>>(
      Gq0, F2, Gsl, SQ, LQ, HM, DHD);
  SM(Gsl, nullptr, (long)LQ * HM, SQ);
  GEMM(H1, WrowQ, nullptr, F1, nullptr, (int)TOKM, DM, DM, DM, DM, DM, 0, 0, 0, 1, 1.f, 0, 0);
  mul_sw_kernel<<<dim3((unsigned)((TOKM * DM + 255) / 256)), dim3(256), 0, stream>>>(
      F1, Gsl, H0, SQ, LQ, HM, DHD, TOKM * DM);
  GEMM(H1, WrowK, nullptr, nullptr, H0 + TOKM * DM, (int)TOKM, DM, DM, DM, DM, DM,
       0, 0, 0, 1, DHI, 0, 0);
  GEMM(H1, WrowV, nullptr, nullptr, H0 + 2 * TOKM * DM, (int)TOKM, DM, DM, DM, DM, DM,
       0, 0, 0, 1, 1.f, 0, 0);
  GEMM(H1, WrowG, BP(ROW_GB), F1, nullptr, (int)TOKM, DM, DM, DM, DM, DM,
       0, 0, 0, 1, 1.f, 2, 0);
  // pack q,k: [s,l,h,d] -> [h, l, (s,d)]
  P16(H0, H2, HM, LQ, SQ, DHD, (long)DHD, (long)HM * DHD, (long)LQ * HM * DHD, 1L,
      (long)LQ * SQ * DHD, (long)SQ * DHD, (long)DHD, 1L);
  P16(H0 + TOKM * DM, H3, HM, LQ, SQ, DHD, (long)DHD, (long)HM * DHD, (long)LQ * HM * DHD, 1L,
      (long)LQ * SQ * DHD, (long)SQ * DHD, (long)DHD, 1L);
  GEMM(H2, H3, nullptr, Gm, nullptr, LQ, LQ, SQ * DHD, SQ * DHD, SQ * DHD, LQ,
       (long)LQ * SQ * DHD, (long)LQ * SQ * DHD, (long)LQ * LQ, HM, 1.f, 0, 0);
  softmax_bias_kernel<<<dim3((unsigned)(((long)HM * LQ + 7) / 8)), dim3(256), 0, stream>>>(
      Gm, Gbias, H2, HM, LQ, 0);
  // pack v: [s,l,h,d] -> [h, (s,d), l]
  P16(H0 + 2 * TOKM * DM, H3, HM, SQ, DHD, LQ,
      (long)DHD, (long)LQ * HM * DHD, 1L, (long)HM * DHD,
      (long)SQ * DHD * LQ, (long)DHD * LQ, (long)LQ, 1L);
  GEMM(H2, H3, nullptr, F2, nullptr, LQ, SQ * DHD, LQ, LQ, LQ, SQ * DHD,
       (long)LQ * LQ, (long)SQ * DHD * LQ, (long)LQ * SQ * DHD, HM, 1.f, 0, 0);
  // unpack [h,q,(s,d)] -> [s,q,h,d]
  P32(F2, F0, SQ, LQ, HM, DHD, (long)DHD, (long)SQ * DHD, (long)LQ * SQ * DHD, 1L,
      (long)LQ * HM * DHD, (long)HM * DHD, (long)DHD, 1L, 0);
  EW(F1, F0, nullptr, H2, TOKM * DM, 2);  // gate * out
  GEMM(H2, WrowO, BP(ROW_OB), MSA, nullptr, (int)TOKM, DM, DM, DM, DM, DM,
       0, 0, 0, 1, 1.f, 0, 1);

  // ---- column attention ----
  LN(MSA, COL_NM_G, COL_NM_B, nullptr, H1, TOKM, DM);
  GEMM(H1, WcolQ, nullptr, nullptr, H0, (int)TOKM, DM, DM, DM, DM, DM, 0, 0, 0, 1, DHI, 0, 0);
  GEMM(H1, WcolK, nullptr, nullptr, H0 + TOKM * DM, (int)TOKM, DM, DM, DM, DM, DM,
       0, 0, 0, 1, 1.f, 0, 0);
  GEMM(H1, WcolV, nullptr, nullptr, H0 + 2 * TOKM * DM, (int)TOKM, DM, DM, DM, DM, DM,
       0, 0, 0, 1, 1.f, 0, 0);
  GEMM(H1, WcolG, BP(COL_GB), F1, nullptr, (int)TOKM, DM, DM, DM, DM, DM,
       0, 0, 0, 1, 1.f, 2, 0);
  // pack q,k: [s,l,h,d] -> [(l,h), s, d]
  P16(H0, H2, LQ, HM, SQ, DHD, (long)HM * DHD, (long)DHD, (long)LQ * HM * DHD, 1L,
      (long)HM * SQ * DHD, (long)SQ * DHD, (long)DHD, 1L);
  P16(H0 + TOKM * DM, H3, LQ, HM, SQ, DHD, (long)HM * DHD, (long)DHD, (long)LQ * HM * DHD, 1L,
      (long)HM * SQ * DHD, (long)SQ * DHD, (long)DHD, 1L);
  {
    _Float16* A16 = H0 + 3 * TOKM * DM;
    for (int c = 0; c < 4; ++c) {
      long boff = (long)c * 64 * HM;  // 512 batches per chunk
      GEMM(H2 + boff * SQ * DHD, H3 + boff * SQ * DHD, nullptr, F2, nullptr,
           SQ, SQ, DHD, DHD, DHD, SQ, (long)SQ * DHD, (long)SQ * DHD, (long)SQ * SQ,
           64 * HM, 1.f, 0, 0);
      SM(F2, A16, (long)64 * HM * SQ, SQ);
      // v chunk: [s,l,h,d] -> [(l',h), d, s]
      P16(H0 + 2 * TOKM * DM + (long)c * 64 * HM * DHD, H4, 64, HM, DHD, SQ,
          (long)HM * DHD, (long)DHD, 1L, (long)LQ * HM * DHD,
          (long)HM * DHD * SQ, (long)DHD * SQ, (long)SQ, 1L);
      GEMM(A16, H4, nullptr, Gm, nullptr, SQ, DHD, SQ, SQ, SQ, DHD,
           (long)SQ * SQ, (long)DHD * SQ, (long)SQ * DHD, 64 * HM, 1.f, 0, 0);
      // scatter [(l',h), s, d] -> F0 [s, l, h, d]
      P32(Gm, F0 + (long)c * 64 * HM * DHD, SQ, 64, HM, DHD,
          (long)DHD, (long)HM * SQ * DHD, (long)SQ * DHD, 1L,
          (long)LQ * HM * DHD, (long)HM * DHD, (long)DHD, 1L, 0);
    }
  }
  EW(F1, F0, nullptr, H2, TOKM * DM, 2);
  GEMM(H2, WcolO, BP(COL_OB), MSA, nullptr, (int)TOKM, DM, DM, DM, DM, DM,
       0, 0, 0, 1, 1.f, 0, 1);

  // ---- msa feed-forward ----
  LN(MSA, MFF_NG, MFF_NB, nullptr, H1, TOKM, DM);
  GEMM(H1, Wff1, BP(MFF_B1), nullptr, H0, (int)TOKM, 4 * DM, DM, DM, DM, 4 * DM,
       0, 0, 0, 1, 1.f, 1, 0);
  GEMM(H0, Wff2, BP(MFF_B2), MSA, nullptr, (int)TOKM, DM, 4 * DM, 4 * DM, 4 * DM, DM,
       0, 0, 0, 1, 1.f, 0, 1);

  // ================= Phase 2: msa2pair =================
  LN(MSA, M2P_NG, M2P_NB, nullptr, H1, TOKM, DM);
  GEMM(H1, WplM, BP(M2P_PLB), nullptr, H2, (int)TOKM, DHD, DM, DM, DM, DHD,
       0, 0, 0, 1, 1.f, 0, 0);
  GEMM(H1, WprM, BP(M2P_PRB), nullptr, H2 + TOKM * DHD, (int)TOKM, DHD, DM, DM, DM, DHD,
       0, 0, 0, 1, 1.f / SQ, 0, 0);
  // pack [(l,i), s] and [(m,j), s]
  P16(H2, H3, LQ, DHD, SQ, 1, (long)DHD, 1L, (long)LQ * DHD, 0L,
      (long)DHD * SQ, (long)SQ, 1L, 0L);
  P16(H2 + TOKM * DHD, H3 + (long)LQ * DHD * SQ, LQ, DHD, SQ, 1,
      (long)DHD, 1L, (long)LQ * DHD, 0L, (long)DHD * SQ, (long)SQ, 1L, 0L);
  {
    _Float16* Lp = H3;
    _Float16* Rp = H3 + (long)LQ * DHD * SQ;
    for (int c = 0; c < 8; ++c) {   // 32 residues per chunk
      const int MR = 32 * DHD;      // 1024 rows
      GEMM(Lp + (long)c * MR * SQ, Rp, nullptr, F2, nullptr, MR, LQ * DHD, SQ,
           SQ, SQ, LQ * DHD, 0, 0, 0, 1, 1.f, 0, 0);
      // [(l,i),(m,j)] -> [(l,m), i*32+j] f16
      P3216(F2, H2, 32, LQ, DHD, DHD,
            (long)DHD * LQ * DHD, (long)DHD, (long)LQ * DHD, 1L,
            (long)LQ * DHD * DHD, (long)DHD * DHD, (long)DHD, 1L);
      GEMM(H2, WpoM, BP(M2P_POB), PAIR + (long)c * 32 * LQ * DP, nullptr,
           32 * LQ, DP, DHD * DHD, DHD * DHD, DHD * DHD, DP, 0, 0, 0, 1, 1.f, 0, 1);
    }
  }

  // ================= Phase 3: pair2pair =================
  // state-gated rbf pair bias -> F1
  LN((const float*)d_in[IN_STATE], P2P_NSG, P2P_NSB, nullptr, SML, LQ, DS);
  GEMM(SML, Wp2L, BP(P2P_PLB), Gq0, nullptr, LQ, DHS, DS, DS, DS, DHS, 0, 0, 0, 1, 1.f, 0, 0);
  GEMM(SML, Wp2R, BP(P2P_PRB), Gst, nullptr, LQ, DHS, DS, DS, DS, DHS, 0, 0, 0, 1, 1.f, 0, 0);
  outer16_kernel<<<dim3((unsigned)((TOKP * 256 + 255) / 256)), dim3(256), 0, stream>>>(
      Gq0, Gst, H0, LQ, DHS, TOKP * 256);
  GEMM(H0, Wtg, BP(P2P_TGB), F1, nullptr, (int)TOKP, DP, DHS * DHS, DHS * DHS, DHS * DHS, DP,
       0, 0, 0, 1, 1.f, 2, 0);
  EW((const float*)d_in[IN_RBF], nullptr, nullptr, H4, TOKP * DR, 0);
  GEMM(H4, Werbf2, BP(P2P_ERBFB), F2, nullptr, (int)TOKP, DP, DR, DR, DR, DP,
       0, 0, 0, 1, 1.f, 0, 0);
  EW(F1, F2, F1, nullptr, TOKP * DP, 2);   // F1 = rbf_b (kept live for both axials)

  // ---- triangle multiply (outgoing / incoming) ----
  auto TRI = [&](int base, bool outgoing, _Float16* Wlp, _Float16* Wlg,
                 _Float16* Wrp, _Float16* Wrg, _Float16* Wg, _Float16* Wop) {
    LN(PAIR, base, base + 1, nullptr, H1, TOKP, DP);
    GEMM(H1, Wlp, BP(base + 3), F2, nullptr, (int)TOKP, DHD, DP, DP, DP, DHD,
         0, 0, 0, 1, 1.f, 0, 0);
    GEMM(H1, Wlg, BP(base + 7), F3, nullptr, (int)TOKP, DHD, DP, DP, DP, DHD,
         0, 0, 0, 1, 1.f, 2, 0);
    EW(F3, F2, nullptr, H2, TOKP * DHD, 2);                 // left
    GEMM(H1, Wrp, BP(base + 5), F2, nullptr, (int)TOKP, DHD, DP, DP, DP, DHD,
         0, 0, 0, 1, 1.f, 0, 0);
    GEMM(H1, Wrg, BP(base + 9), F3, nullptr, (int)TOKP, DHD, DP, DP, DP, DHD,
         0, 0, 0, 1, 1.f, 2, 0);
    EW(F3, F2, nullptr, H2 + TOKP * DHD, TOKP * DHD, 2);    // right
    // per-channel packs [d, i, k] / [d, j, k]
    long sik = outgoing ? (long)LQ * DHD : (long)DHD;
    long skk = outgoing ? (long)DHD : (long)LQ * DHD;
    P16(H2, H3, DHD, LQ, LQ, 1, 1L, sik, skk, 0L, (long)LQ * LQ, (long)LQ, 1L, 0L);
    P16(H2 + TOKP * DHD, H3 + TOKP * DHD, DHD, LQ, LQ, 1, 1L, sik, skk, 0L,
        (long)LQ * LQ, (long)LQ, 1L, 0L);
    GEMM(H3, H3 + TOKP * DHD, nullptr, Gm, nullptr, LQ, LQ, LQ, LQ, LQ, LQ,
         (long)LQ * LQ, (long)LQ * LQ, (long)LQ * LQ, DHD, 1.f, 0, 0);
    // [d,i,j] -> [(i,j), d]
    P32(Gm, F2, LQ, LQ, DHD, 1, (long)LQ, 1L, (long)LQ * LQ, 0L,
        (long)LQ * DHD, (long)DHD, 1L, 0L, 0);
    LN(F2, base + 12, base + 13, nullptr, H2, TOKP, DHD);
    GEMM(H2, Wop, BP(base + 15), F0, nullptr, (int)TOKP, DP, DHD, DHD, DHD, DP,
         0, 0, 0, 1, 1.f, 0, 0);
    GEMM(H1, Wg, BP(base + 11), F3, nullptr, (int)TOKP, DP, DP, DP, DP, DP,
         0, 0, 0, 1, 1.f, 2, 0);
    EW(F3, F0, PAIR, nullptr, TOKP * DP, 4);  // pair += gate * out
  };
  TRI(TO_NG, true,  WtoLP, WtoLG, WtoRP, WtoRG, WtoG, WtoOP);
  TRI(TI_NG, false, WtiLP, WtiLG, WtiRP, WtiRG, WtiG, WtiOP);

  // ---- biased axial attention (tied over the other axis) ----
  auto AXIAL = [&](int base, bool is_row, _Float16* Wq, _Float16* Wk, _Float16* Wv,
                   _Float16* Wg, _Float16* Wb, _Float16* Wo) {
    // bias head projection of LN(rbf_b)
    LN(F1, base + 9, base + 10, nullptr, H4, TOKP, DP);
    float* bb = Gm + 262144;
    GEMM(H4, Wb, nullptr, bb, nullptr, (int)TOKP, HP, DP, DP, DP, HP, 0, 0, 0, 1, 1.f, 0, 0);
    LN(PAIR, base, base + 1, nullptr, H1, TOKP, DP);
    GEMM(H1, Wq, nullptr, nullptr, H0, (int)TOKP, DP, DP, DP, DP, DP, 0, 0, 0, 1, DHI, 0, 0);
    GEMM(H1, Wk, nullptr, nullptr, H0 + TOKP * DP, (int)TOKP, DP, DP, DP, DP, DP,
         0, 0, 0, 1, 1.f / LQ, 0, 0);
    GEMM(H1, Wv, nullptr, nullptr, H0 + 2 * TOKP * DP, (int)TOKP, DP, DP, DP, DP, DP,
         0, 0, 0, 1, 1.f, 0, 0);
    GEMM(H1, Wg, BP(base + 6), F3, nullptr, (int)TOKP, DP, DP, DP, DP, DP,
         0, 0, 0, 1, 1.f, 2, 0);
    // pack q,k: [h, i, (n,k)]
    long si = is_row ? (long)LQ * HP * DHD : (long)HP * DHD;
    long sn = is_row ? (long)HP * DHD : (long)LQ * HP * DHD;
    P16(H0, H2, HP, LQ, LQ, DHD, (long)DHD, si, sn, 1L,
        (long)LQ * LQ * DHD, (long)LQ * DHD, (long)DHD, 1L);
    P16(H0 + TOKP * DP, H3, HP, LQ, LQ, DHD, (long)DHD, si, sn, 1L,
        (long)LQ * LQ * DHD, (long)LQ * DHD, (long)DHD, 1L);
    GEMM(H2, H3, nullptr, Gm, nullptr, LQ, LQ, LQ * DHD, LQ * DHD, LQ * DHD, LQ,
         (long)LQ * LQ * DHD, (long)LQ * LQ * DHD, (long)LQ * LQ, HP, 1.f, 0, 0);
    softmax_bias_kernel<<<dim3((unsigned)(((long)HP * LQ + 7) / 8)), dim3(256), 0, stream>>>(
        Gm, bb, SML, HP, LQ, is_row ? 1 : 0);
    // pack v: [h, (n,d), j]
    P16(H0 + 2 * TOKP * DP, H3, HP, LQ, DHD, LQ, (long)DHD, sn, 1L, si,
        (long)LQ * DHD * LQ, (long)DHD * LQ, (long)LQ, 1L);
    GEMM(SML, H3, nullptr, F2, nullptr, LQ, LQ * DHD, LQ, LQ, LQ, LQ * DHD,
         (long)LQ * LQ, (long)LQ * DHD * LQ, (long)LQ * LQ * DHD, HP, 1.f, 0, 0);
    // [h, i, (n,d)] -> token-major [tok, h*DH+d]
    P32(F2, F0, HP, LQ, LQ, DHD, (long)LQ * LQ * DHD, (long)LQ * DHD, (long)DHD, 1L,
        (long)DHD, si, sn, 1L, 0);
    EW(F3, F0, nullptr, H2, TOKP * DP, 2);
    GEMM(H2, Wo, BP(base + 8), PAIR, nullptr, (int)TOKP, DP, DP, DP, DP, DP,
         0, 0, 0, 1, 1.f, 0, 1);
  };
  AXIAL(PR_NM_G, true,  WprQ2, WprK2, WprV2, WprG2, WprB2, WprO2);
  AXIAL(PC_NM_G, false, WpcQ2, WpcK2, WpcV2, WpcG2, WpcB2, WpcO2);

  // ---- pair feed-forward ----
  LN(PAIR, PFF_NG, PFF_NB, nullptr, H1, TOKP, DP);
  GEMM(H1, Wpf1, BP(PFF_B1), nullptr, H0, (int)TOKP, 2 * DP, DP, DP, DP, 2 * DP,
       0, 0, 0, 1, 1.f, 1, 0);
  GEMM(H0, Wpf2, BP(PFF_B2), PAIR, nullptr, (int)TOKP, DP, 2 * DP, 2 * DP, 2 * DP, DP,
       0, 0, 0, 1, 1.f, 0, 1);

  // ----- outputs: (msa, pair) concatenated flat -----
  hipMemcpyAsync(d_out, MSA, TOKM * DM * sizeof(float), hipMemcpyDeviceToDevice, stream);
  hipMemcpyAsync((float*)d_out + TOKM * DM, PAIR, TOKP * DP * sizeof(float),
                 hipMemcpyDeviceToDevice, stream);
}
